// BiAttention_40819369181436
// MI455X (gfx1250) — matrix operand order repaired and checked by reading, not compiled, not run on hardware
//
#include <hip/hip_runtime.h>
#include <hip/hip_bf16.h>

// ---------- types ----------
typedef __attribute__((ext_vector_type(16))) __bf16        v16bf;
typedef __attribute__((ext_vector_type(8)))  float         v8f;
typedef __attribute__((ext_vector_type(16))) unsigned short u16x16;
typedef __attribute__((ext_vector_type(8)))  unsigned short u16x8;
typedef __attribute__((ext_vector_type(4)))  unsigned int  u32x4;
typedef __attribute__((ext_vector_type(8)))  int           i32x8;
typedef __attribute__((ext_vector_type(4)))  int           i32x4;

// ---------- CDNA5 feature probes ----------
#if __has_builtin(__builtin_amdgcn_global_load_async_to_lds_b128)
#define HAVE_ASYNC_LDS 1
#else
#define HAVE_ASYNC_LDS 0
#endif

#if __has_builtin(__builtin_amdgcn_tensor_load_to_lds) && \
    __has_builtin(__builtin_amdgcn_s_wait_tensorcnt)
#define HAVE_TDM 1
#else
#define HAVE_TDM 0
#endif

#if HAVE_ASYNC_LDS
// Builtin params are typed v4i pointers: AS1 (prints as __device__) src,
// AS3 LDS dst. Build them by integer round-trip: global generic == AS1
// numerically; LDS offset = generic addr[31:0] (ISA 10.2).
#define GASYNC_B128(gp, lp)                                                   \
  __builtin_amdgcn_global_load_async_to_lds_b128(                             \
      (__attribute__((address_space(1))) i32x4*)(size_t)(gp),                 \
      (__attribute__((address_space(3))) i32x4*)(unsigned)(size_t)(lp),       \
      0, 0)
#endif

__device__ __forceinline__ void wait_async0() {
#if __has_builtin(__builtin_amdgcn_s_wait_asynccnt)
  __builtin_amdgcn_s_wait_asynccnt(0);
#else
  asm volatile("s_wait_asynccnt 0x0" ::: "memory");
#endif
}

// fp32 -> bf16, round-to-nearest-even
__device__ __forceinline__ unsigned short f2bf(float f) {
  unsigned u = __builtin_bit_cast(unsigned, f);
  u += 0x7fffu + ((u >> 16) & 1u);
  return (unsigned short)(u >> 16);
}

// A-fragment (16x32 bf16, M rows contiguous in K):
// lane L: m = L%16, elements 0..15 = K kb..kb+15, kb = (L/16)*16
__device__ __forceinline__ v16bf load_afrag(const unsigned short* row, int kk, int lane) {
  int kb = kk + ((lane >> 4) << 4);
  return __builtin_bit_cast(v16bf, *(const u16x16*)(row + kb));
}

// B-fragment (32x16 bf16) from TRANSPOSED storage (row = N column, contiguous in K):
// lane L: n = L%16, elements 0..15 = K kb..kb+15, kb = (L/16)*16
__device__ __forceinline__ v16bf load_bfrag(const unsigned short* colrow, int kk, int lane) {
  int kb = kk + ((lane >> 4) << 4);
  return __builtin_bit_cast(v16bf, *(const u16x16*)(colrow + kb));
}

#define WMMA_BF16(A, B, C) \
  __builtin_amdgcn_wmma_f32_16x16x32_bf16(false, (A), false, (B), (short)0, (C), false, false)

// ---------------------------------------------------------------------------
// K0: pack conv weights OIHW fp32 -> bf16 [tap][co][ci]; fold BN into scale/shift
// ---------------------------------------------------------------------------
__global__ __launch_bounds__(256) void pack_weights(
    const float* __restrict__ ck, const float* __restrict__ bnw,
    const float* __restrict__ bnb, const float* __restrict__ bnm,
    const float* __restrict__ bnv, unsigned short* __restrict__ Wp,
    float* __restrict__ scale, float* __restrict__ shift) {
  int idx = blockIdx.x * 256 + threadIdx.x;            // 256*256*9 = 589824
  int kw = idx % 3; int r = idx / 3;
  int kh = r % 3;   r /= 3;
  int ci = r % 256; int co = r / 256;
  Wp[((kh * 3 + kw) * 256 + co) * 256 + ci] = f2bf(ck[idx]);
  if (blockIdx.x == 0) {
    int c = threadIdx.x;
    float inv = bnw[c] * rsqrtf(bnv[c] + 1e-5f);
    scale[c] = inv;
    shift[c] = bnb[c] - bnm[c] * inv;
  }
}

// ---------------------------------------------------------------------------
// K1: pooled mean + Linear(C->C); output TRANSPOSED bf16: Yt[n][co][j], j = h or w
// ---------------------------------------------------------------------------
template <int DIR>
__global__ __launch_bounds__(256) void pool_proj(
    const float* __restrict__ x, const float* __restrict__ wgt,
    const float* __restrict__ bias, unsigned short* __restrict__ Yt) {
  int n = blockIdx.x >> 6;
  int j = blockIdx.x & 63;
  __shared__ float pm[256];
  int c = threadIdx.x;
  float s = 0.f;
  if (DIR == 0) {
    const float* p = x + (((size_t)(n * 256 + c) * 64 + j) * 64);
#pragma unroll 8
    for (int i = 0; i < 64; ++i) s += p[i];
  } else {
    const float* p = x + ((size_t)(n * 256 + c) * 64 * 64 + j);
#pragma unroll 8
    for (int i = 0; i < 64; ++i) s += p[i * 64];
  }
  pm[c] = s * (1.f / 64.f);
  __syncthreads();
  int co = c;
  float acc = bias[co];
  const float* wr = wgt + (size_t)co * 256;
#pragma unroll 8
  for (int cc = 0; cc < 256; ++cc) acc += pm[cc] * wr[cc];
  Yt[((size_t)n * 256 + co) * 64 + j] = f2bf(acc);
}

// ---------------------------------------------------------------------------
// K2: axial attention per slice (512 blocks, 8 waves).
//   S = X(256x64) * Y(64x256); softmax rows; O = P(256x256) * X(256x64)
// DIR 0 (H): slice (n,w), j = h, writes Oh (fp32 NCHW).
// DIR 1 (W): slice (n,h), j = w, fuses out = gamma*(Oh+Ow)+x -> bf16 NHWC.
// Reproduces the torch .repeat quirk: y index = blockIdx % N.
// ---------------------------------------------------------------------------
template <int DIR>
__global__ __launch_bounds__(256) void attn_kernel(
    const float* __restrict__ x, const unsigned short* __restrict__ Yg,
    float* __restrict__ Oh, unsigned short* __restrict__ outb,
    const float* __restrict__ gammaPtr) {
  __shared__ __align__(32) unsigned short Xr[256 * 64];   // X row-major  (c, j)
  __shared__ __align__(32) unsigned short Xt[64 * 256];   // X transposed (j, c)
  __shared__ __align__(32) unsigned short Yt[256 * 64];   // Y transposed (co, j)
  __shared__ __align__(32) float          Sb[64 * 256];   // S row-block fp32
  __shared__ __align__(32) unsigned short Pb[64 * 256];   // softmax(S) bf16
  __shared__ float redm[64 * 4];
  __shared__ float reds[64 * 4];

  const int b = blockIdx.x;
  const int n = b >> 6;
  const int sl = b & 63;            // w (DIR0) or h (DIR1)
  const int yIdx = b & 7;           // b % N quirk
  const int tid = threadIdx.x;
  const int wid = tid >> 5, lane = tid & 31;
  const float g = gammaPtr[0];

  __builtin_prefetch(Yg + (size_t)yIdx * 16384 + tid * 64, 0, 1);

  // ---- stage Y (bf16, 32KB contiguous): TDM -> async-LDS -> dword copy ----
#if HAVE_TDM
  if (tid < 32) {
    unsigned ldsa = (unsigned)(size_t)(void*)Yt;
    unsigned long long ga = (unsigned long long)(size_t)(Yg + (size_t)yIdx * 16384);
    u32x4 g0;
    g0.x = 1u;                                   // count=1 valid descriptor
    g0.y = ldsa;                                 // lds_addr
    g0.z = (unsigned)ga;                         // global_addr[31:0]
    g0.w = (unsigned)(ga >> 32) | (2u << 30);    // global_addr[56:32] | type=2
    i32x8 g1;
    g1[0] = (int)(1u << 16);                     // data_size = 1 (2 bytes)
    g1[1] = (int)(16384u << 16);                 // tensor_dim0[15:0]
    g1[2] = (int)(1u << 16);                     // tensor_dim0[31:16]=0 | tensor_dim1=1
    g1[3] = (int)(16384u << 16);                 // tile_dim0 = 16384
    g1[4] = 1;                                   // tile_dim1 = 1
    g1[5] = 16384;                               // tensor_dim0_stride
    g1[6] = 0; g1[7] = 0;
    i32x4 z4 = {};
#if (__clang_major__ >= 23)
    i32x8 z8 = {};
    __builtin_amdgcn_tensor_load_to_lds(g0, g1, z4, z4, z8, 0);   // 6-arg form
#else
    __builtin_amdgcn_tensor_load_to_lds(g0, g1, z4, z4, 0);       // 5-arg form
#endif
    __builtin_amdgcn_s_wait_tensorcnt(0);
  }
#elif HAVE_ASYNC_LDS
  {
    const unsigned short* src = Yg + (size_t)yIdx * 16384;
#pragma unroll
    for (int k = 0; k < 8; ++k) {
      int q = k * 256 + tid;                     // 2048 chunks of 16B
      GASYNC_B128(src + q * 8, Yt + q * 8);
    }
    wait_async0();
  }
#else
  {
    const unsigned* src = (const unsigned*)(Yg + (size_t)yIdx * 16384);
    unsigned* dst = (unsigned*)Yt;
#pragma unroll
    for (int k = 0; k < 32; ++k) dst[tid + k * 256] = src[tid + k * 256];
  }
#endif

  // ---- stage X (fp32 -> bf16, both orientations) ----
  {
    int c = tid;
#pragma unroll 4
    for (int j = 0; j < 64; ++j) {
      size_t addr = (DIR == 0)
        ? (((size_t)(n * 256 + c) * 64 + j) * 64 + sl)
        : (((size_t)(n * 256 + c) * 64 + sl) * 64 + j);
      unsigned short v = f2bf(x[addr]);
      Xr[c * 64 + j] = v;
      Xt[j * 256 + c] = v;
    }
  }
  __syncthreads();

  for (int rb = 0; rb < 4; ++rb) {
    // ---- S GEMM: 64x256, K=64.  wave: mt = wid&3 (16 rows), 8 n-tiles ----
    {
      int mt = wid & 3;
      int ntB = (wid >> 2) * 8;
      v8f acc[8];
#pragma unroll
      for (int i = 0; i < 8; ++i) acc[i] = (v8f)(0.f);
#pragma unroll
      for (int kk = 0; kk < 64; kk += 32) {
        v16bf a = load_afrag(Xr + (rb * 64 + mt * 16 + (lane & 15)) * 64, kk, lane);
#pragma unroll
        for (int nt = 0; nt < 8; ++nt) {
          v16bf bf = load_bfrag(Yt + ((ntB + nt) * 16 + (lane & 15)) * 64, kk, lane);
          acc[nt] = WMMA_BF16(a, bf, acc[nt]);
        }
      }
#pragma unroll
      for (int nt = 0; nt < 8; ++nt)
#pragma unroll
        for (int v = 0; v < 8; ++v) {
          int ml = mt * 16 + v + 8 * (lane >> 4);
          int col = (ntB + nt) * 16 + (lane & 15);
          Sb[ml * 256 + col] = acc[nt][v];
        }
    }
    __syncthreads();

    // ---- softmax over 256 cols, 4 threads per row ----
    {
      int r = tid & 63, p = tid >> 6;
      float* base = Sb + r * 256 + p * 64;
      float m = -3.4e38f;
#pragma unroll 8
      for (int i = 0; i < 64; ++i) m = fmaxf(m, base[i]);
      redm[r * 4 + p] = m;
      __syncthreads();
      float rm = fmaxf(fmaxf(redm[r * 4], redm[r * 4 + 1]),
                       fmaxf(redm[r * 4 + 2], redm[r * 4 + 3]));
      float s = 0.f;
#pragma unroll 8
      for (int i = 0; i < 64; ++i) {
        float e = __expf(base[i] - rm);
        base[i] = e;
        s += e;
      }
      reds[r * 4 + p] = s;
      __syncthreads();
      float inv = 1.f / (reds[r * 4] + reds[r * 4 + 1] + reds[r * 4 + 2] + reds[r * 4 + 3]);
      unsigned short* pb = Pb + r * 256 + p * 64;
#pragma unroll 8
      for (int i = 0; i < 64; ++i) pb[i] = f2bf(base[i] * inv);
    }
    __syncthreads();

    // ---- O GEMM: 64x64, K=256.  wave: mt2 = wid&3, 2 n-tiles ----
    {
      int mt2 = wid & 3;
      int ntB2 = (wid >> 2) * 2;
      v8f acc2[2];
      acc2[0] = (v8f)(0.f);
      acc2[1] = (v8f)(0.f);
#pragma unroll
      for (int kk = 0; kk < 256; kk += 32) {
        v16bf a = load_afrag(Pb + (mt2 * 16 + (lane & 15)) * 256, kk, lane);
#pragma unroll
        for (int t = 0; t < 2; ++t) {
          v16bf bf = load_bfrag(Xt + ((ntB2 + t) * 16 + (lane & 15)) * 256, kk, lane);
          acc2[t] = WMMA_BF16(a, bf, acc2[t]);
        }
      }
#pragma unroll
      for (int t = 0; t < 2; ++t)
#pragma unroll
        for (int v = 0; v < 8; ++v) {
          int c = rb * 64 + mt2 * 16 + v + 8 * (lane >> 4);
          int j = (ntB2 + t) * 16 + (lane & 15);
          float val = acc2[t][v];
          if (DIR == 0) {
            Oh[((size_t)(n * 256 + c) * 64 + j) * 64 + sl] = val;
          } else {
            size_t addr = ((size_t)(n * 256 + c) * 64 + sl) * 64 + j;   // NCHW reads
            float o = g * (Oh[addr] + val) + x[addr];
            outb[(((size_t)n * 64 + sl) * 64 + j) * 256 + c] = f2bf(o); // NHWC write
          }
        }
    }
    __syncthreads();
  }
}

// ---------------------------------------------------------------------------
// K3: 3x3 conv as implicit GEMM (M=co64, N=64 pixels (8x8), K=256*9) + BN + ReLU
// Input u is NHWC bf16 -> halo rows are contiguous 512B: async copy to LDS.
// ---------------------------------------------------------------------------
__global__ __launch_bounds__(256) void conv_bn_relu(
    const unsigned short* __restrict__ u, const unsigned short* __restrict__ Wp,
    const float* __restrict__ scale, const float* __restrict__ shift,
    float* __restrict__ y) {
  __shared__ __align__(32) unsigned short uh[100 * 256];  // 10x10 halo x ci, bf16

  const int bid = blockIdx.x;            // 2048 = 512 pixel tiles * 4 co tiles
  const int cot = bid & 3;
  const int pt = bid >> 2;
  const int n = pt >> 6;
  const int ptl = pt & 63;
  const int ph0 = (ptl >> 3) * 8, pw0 = (ptl & 7) * 8;
  const int tid = threadIdx.x;

  // ---- stage halo (zero padded): async gather of 512B rows when in-bounds ----
#if HAVE_ASYNC_LDS
  for (int it = 0; it < 13; ++it) {
    int q = it * 256 + tid;                   // 100 rows * 32 chunks = 3200
    if (q < 3200) {
      int row = q >> 5, ch = q & 31;          // 16B chunk within a halo row
      int h = row / 10, w = row % 10;
      int hh = ph0 + h - 1, wwp = pw0 + w - 1;
      unsigned short* ldsp = uh + row * 256 + ch * 8;
      if (hh >= 0 && hh < 64 && wwp >= 0 && wwp < 64) {
        const unsigned short* gp =
            u + (((size_t)(n * 64 + hh)) * 64 + wwp) * 256 + ch * 8;
        GASYNC_B128(gp, ldsp);
      } else {
        u16x8 z = {};
        *(u16x8*)ldsp = z;
      }
    }
  }
  wait_async0();
#else
  for (int it = 0; it < 100; ++it) {
    int h = it / 10, w = it % 10;
    int hh = ph0 + h - 1, wwp = pw0 + w - 1;
    unsigned short v = 0;
    if (hh >= 0 && hh < 64 && wwp >= 0 && wwp < 64)
      v = u[(((size_t)(n * 64 + hh)) * 64 + wwp) * 256 + tid];
    uh[it * 256 + tid] = v;
  }
#endif
  __syncthreads();

  const int wid = tid >> 5, lane = tid & 31;
  const int mt = wid & 3;
  const int ntB = (wid >> 2) * 2;
  v8f acc[2];
  acc[0] = (v8f)(0.f);
  acc[1] = (v8f)(0.f);
  const int coRow = cot * 64 + mt * 16 + (lane & 15);

  for (int tap = 0; tap < 9; ++tap) {
    const int kh = tap / 3, kw = tap % 3;
    const unsigned short* wrow = Wp + ((size_t)(tap * 256 + coRow)) * 256;
#pragma unroll
    for (int kc = 0; kc < 256; kc += 32) {
      v16bf a = load_afrag(wrow, kc, lane);
#pragma unroll
      for (int t = 0; t < 2; ++t) {
        int p = (ntB + t) * 16 + (lane & 15);
        int hi = (p >> 3) + kh, wi = (p & 7) + kw;
        v16bf bf = load_bfrag(uh + (hi * 10 + wi) * 256, kc, lane);
        acc[t] = WMMA_BF16(a, bf, acc[t]);
      }
    }
  }

#pragma unroll
  for (int t = 0; t < 2; ++t)
#pragma unroll
    for (int v = 0; v < 8; ++v) {
      int co = cot * 64 + mt * 16 + v + 8 * (lane >> 4);
      int p = (ntB + t) * 16 + (lane & 15);
      float val = acc[t][v] * scale[co] + shift[co];
      val = fmaxf(val, 0.f);
      y[((size_t)(n * 256 + co) * 64 + ph0 + (p >> 3)) * 64 + pw0 + (p & 7)] = val;
    }
}

// ---------------------------------------------------------------------------
extern "C" void kernel_launch(void* const* d_in, const int* in_sizes, int n_in,
                              void* d_out, int out_size, void* d_ws, size_t ws_size,
                              hipStream_t stream) {
  const float* x    = (const float*)d_in[0];
  const float* wh   = (const float*)d_in[1];
  const float* bh   = (const float*)d_in[2];
  const float* ww   = (const float*)d_in[3];
  const float* bw   = (const float*)d_in[4];
  const float* ck   = (const float*)d_in[5];
  const float* bnw  = (const float*)d_in[6];
  const float* bnb  = (const float*)d_in[7];
  const float* bnm  = (const float*)d_in[8];
  const float* bnv  = (const float*)d_in[9];
  const float* gamma = (const float*)d_in[10];

  char* ws = (char*)d_ws;
  size_t o = 0;
  unsigned short* Yh = (unsigned short*)(ws + o); o += (size_t)8 * 256 * 64 * 2;
  unsigned short* Yw = (unsigned short*)(ws + o); o += (size_t)8 * 256 * 64 * 2;
  unsigned short* Wp = (unsigned short*)(ws + o); o += (size_t)9 * 256 * 256 * 2;
  float* scale = (float*)(ws + o); o += 256 * 4;
  float* shift = (float*)(ws + o); o += 256 * 4;
  o = (o + 255) & ~(size_t)255;
  float* Oh = (float*)(ws + o);            o += (size_t)8 * 256 * 64 * 64 * 4;
  unsigned short* outb = (unsigned short*)(ws + o);

  float* y = (float*)d_out;

  pack_weights<<<2304, 256, 0, stream>>>(ck, bnw, bnb, bnm, bnv, Wp, scale, shift);
  pool_proj<0><<<512, 256, 0, stream>>>(x, wh, bh, Yh);
  pool_proj<1><<<512, 256, 0, stream>>>(x, ww, bw, Yw);
  attn_kernel<0><<<512, 256, 0, stream>>>(x, Yh, Oh, nullptr, gamma);
  attn_kernel<1><<<512, 256, 0, stream>>>(x, Yw, Oh, outb, gamma);
  conv_bn_relu<<<2048, 256, 0, stream>>>(outb, Wp, scale, shift, y);
}
